// GraphSAGEWithAutoencoder_15023795601936
// MI455X (gfx1250) — compile-verified, matrix operand order checked by reading, not run on hardware
//
#include <hip/hip_runtime.h>
#include <hip/hip_bf16.h>

// CDNA5 / gfx1250 WMMA types (wave32)
typedef __attribute__((ext_vector_type(16))) __bf16 v16bf;
typedef __attribute__((ext_vector_type(8)))  float  v8f;

#define DEV_INLINE __device__ __forceinline__

// ---------------------------------------------------------------------------
// Weight packing: f32 [K][NOUT] row-major -> per-lane WMMA B-fragments (bf16).
// Fragment order: frag[((ct*KT + ks)*32 + lane)][16], so the update kernel
// loads one aligned 32-byte v16bf per (ct,ks).
// B-matrix per-lane K striping mirrors the documented 16-bit A layout:
//   lanes 0-15: halves 0..7 = K kbase+0..7,  halves 8..15 = K kbase+16..23 (kbase=0)
//   lanes 16-31: same with kbase=8.
// ---------------------------------------------------------------------------
template<int K, int NOUT>
__global__ void pack_w_kernel(const float* __restrict__ w, __bf16* __restrict__ frag) {
    constexpr int KT = K / 32;
    constexpr int NT = NOUT / 16;
    int t = blockIdx.x * blockDim.x + threadIdx.x;
    if (t >= NT * KT * 32) return;
    int lane = t & 31;
    int ks   = (t >> 5) % KT;
    int ct   = t / (32 * KT);
    int n    = 16 * ct + (lane & 15);
    int kb   = (lane >> 4) ? 8 : 0;
    __bf16* d = frag + (size_t)t * 16;
#pragma unroll
    for (int i = 0; i < 8; ++i) {
        d[i]     = (__bf16)w[(size_t)(32 * ks + kb + i)      * NOUT + n];
        d[8 + i] = (__bf16)w[(size_t)(32 * ks + kb + 16 + i) * NOUT + n];
    }
}

// ---------------------------------------------------------------------------
// Encoder: per-node MLP 2 -> 32 -> 32 -> 32, ReLU. K=2 entry makes this a
// VALU job; weights are thread-uniform so they lower to scalar loads.
// ---------------------------------------------------------------------------
__global__ void __launch_bounds__(256)
encoder_kernel(const float* __restrict__ x,
               const float* __restrict__ w1, const float* __restrict__ b1,
               const float* __restrict__ w2, const float* __restrict__ b2,
               const float* __restrict__ w3, const float* __restrict__ b3,
               float* __restrict__ h0, int n) {
    int i = blockIdx.x * blockDim.x + threadIdx.x;
    if (i >= n) return;
    float x0 = x[2 * i], x1 = x[2 * i + 1];
    float t[32], u[32];
#pragma unroll
    for (int j = 0; j < 32; ++j)
        t[j] = fmaxf(fmaf(x1, w1[32 + j], fmaf(x0, w1[j], b1[j])), 0.0f);
#pragma unroll
    for (int j = 0; j < 32; ++j) u[j] = b2[j];
    for (int k = 0; k < 32; ++k) {
        float v = t[k];
#pragma unroll
        for (int j = 0; j < 32; ++j) u[j] = fmaf(v, w2[k * 32 + j], u[j]);
    }
#pragma unroll
    for (int j = 0; j < 32; ++j) { u[j] = fmaxf(u[j], 0.0f); t[j] = b3[j]; }
    for (int k = 0; k < 32; ++k) {
        float v = u[k];
#pragma unroll
        for (int j = 0; j < 32; ++j) t[j] = fmaf(v, w3[k * 32 + j], t[j]);
    }
#pragma unroll
    for (int j = 0; j < 32; ++j) h0[(size_t)i * 32 + j] = fmaxf(t[j], 0.0f);
}

// ---------------------------------------------------------------------------
// Degree: one f32 atomic add per edge (reused by all three layers).
// ---------------------------------------------------------------------------
__global__ void __launch_bounds__(256)
degree_kernel(const int* __restrict__ ei, float* __restrict__ cnt, int n_edges) {
    int e = blockIdx.x * blockDim.x + threadIdx.x;
    if (e >= n_edges) return;
    int d = ei[n_edges + e];
    __hip_atomic_fetch_add(&cnt[d], 1.0f, __ATOMIC_RELAXED, __HIP_MEMORY_SCOPE_AGENT);
}

// ---------------------------------------------------------------------------
// Edge scatter: thread = (edge, 4-feature chunk). Coalesced float4 gather of
// the source row (L2-resident: node tables are <= 25.6 MB << 192 MB L2),
// then 4 non-returning agent-scope f32 atomics into the accumulator.
// ---------------------------------------------------------------------------
template<int D>
__global__ void __launch_bounds__(256)
scatter_kernel(const int* __restrict__ ei, const float* __restrict__ h,
               float* __restrict__ agg, int n_edges) {
    constexpr int CH = D / 4;
    long long tid = (long long)blockIdx.x * blockDim.x + threadIdx.x;
    int e = (int)(tid / CH);
    int c = (int)(tid % CH);
    if (e >= n_edges) return;
    int s = ei[e];
    int d = ei[n_edges + e];
    const float4 v = *((const float4*)(h + (size_t)s * D) + c);
    float* o = agg + (size_t)d * D + 4 * c;
    __hip_atomic_fetch_add(o + 0, v.x, __ATOMIC_RELAXED, __HIP_MEMORY_SCOPE_AGENT);
    __hip_atomic_fetch_add(o + 1, v.y, __ATOMIC_RELAXED, __HIP_MEMORY_SCOPE_AGENT);
    __hip_atomic_fetch_add(o + 2, v.z, __ATOMIC_RELAXED, __HIP_MEMORY_SCOPE_AGENT);
    __hip_atomic_fetch_add(o + 3, v.w, __ATOMIC_RELAXED, __HIP_MEMORY_SCOPE_AGENT);
}

// ---------------------------------------------------------------------------
// SAGE update: out = relu(mean_agg @ wl + bl + h @ wr), one wave per 16-node
// tile using v_wmma_f32_16x16x32_bf16. A-fragments built per the documented
// 16-bit A layout; C/D layout: N = lane&15, M = vgpr + 8*(lane>=16).
// ---------------------------------------------------------------------------
DEV_INLINE v16bf load_a_frag(const float* __restrict__ row, int kb, float s) {
    const float4* p = (const float4*)(row + kb);
    const float4* q = (const float4*)(row + kb + 16);
    float4 p0 = p[0], p1 = p[1], q0 = q[0], q1 = q[1];
    v16bf a;
    a[0]  = (__bf16)(p0.x * s); a[1]  = (__bf16)(p0.y * s);
    a[2]  = (__bf16)(p0.z * s); a[3]  = (__bf16)(p0.w * s);
    a[4]  = (__bf16)(p1.x * s); a[5]  = (__bf16)(p1.y * s);
    a[6]  = (__bf16)(p1.z * s); a[7]  = (__bf16)(p1.w * s);
    a[8]  = (__bf16)(q0.x * s); a[9]  = (__bf16)(q0.y * s);
    a[10] = (__bf16)(q0.z * s); a[11] = (__bf16)(q0.w * s);
    a[12] = (__bf16)(q1.x * s); a[13] = (__bf16)(q1.y * s);
    a[14] = (__bf16)(q1.z * s); a[15] = (__bf16)(q1.w * s);
    return a;
}

template<int K, int NOUT>
__global__ void __launch_bounds__(128)
sage_update_kernel(const float* __restrict__ agg, const float* __restrict__ cnt,
                   const float* __restrict__ h,
                   const __bf16* __restrict__ wlf,  // packed fragments
                   const float* __restrict__ bl,
                   const __bf16* __restrict__ wrf,  // packed fragments
                   float* __restrict__ out, int n_tiles) {
    constexpr int KT = K / 32;
    constexpr int NT = NOUT / 16;
    const int lane = threadIdx.x & 31;
    const int wave = blockIdx.x * (blockDim.x >> 5) + (threadIdx.x >> 5);
    if (wave >= n_tiles) return;                    // wave-uniform: EXEC all-1s below
    const int row_base = wave * 16;
    const int m   = lane & 15;
    const int hi  = lane >> 4;
    const int kb  = hi ? 8 : 0;
    const int row = row_base + m;
    const float s = 1.0f / fmaxf(cnt[row], 1.0f);   // mean aggregation scale

    v16bf a_agg[KT], a_h[KT];
    const float* ar = agg + (size_t)row * K;
    const float* hr = h   + (size_t)row * K;
#pragma unroll
    for (int ks = 0; ks < KT; ++ks) {
        a_agg[ks] = load_a_frag(ar + 32 * ks, kb, s);
        a_h[ks]   = load_a_frag(hr + 32 * ks, kb, 1.0f);
    }

#pragma unroll
    for (int ct = 0; ct < NT; ++ct) {
        const int n = 16 * ct + m;
        const float b0 = bl[n];
        v8f c;
#pragma unroll
        for (int j = 0; j < 8; ++j) c[j] = b0;
#pragma unroll
        for (int ks = 0; ks < KT; ++ks) {
            const size_t fidx = (size_t)((ct * KT + ks) * 32 + lane);
            v16bf bL = *((const v16bf*)wlf + fidx);
            v16bf bR = *((const v16bf*)wrf + fidx);
            c = __builtin_amdgcn_wmma_f32_16x16x32_bf16(false, a_agg[ks], false, bL,
                                                        (short)0, c, false, false);
            c = __builtin_amdgcn_wmma_f32_16x16x32_bf16(false, a_h[ks],   false, bR,
                                                        (short)0, c, false, false);
        }
        float* orow = out + (size_t)(row_base + hi * 8) * NOUT + n;
#pragma unroll
        for (int j = 0; j < 8; ++j)
            orow[(size_t)j * NOUT] = fmaxf(c[j], 0.0f);   // fused ReLU
    }
}

// ---------------------------------------------------------------------------
// Global mean pool: lane = feature (32 == wave width), each wave reads whole
// 128B rows (fully coalesced), partial sums flushed with one atomic per lane.
// ---------------------------------------------------------------------------
__global__ void __launch_bounds__(256)
pool_kernel(const float* __restrict__ h, float* __restrict__ g, int n) {
    int lane  = threadIdx.x & 31;
    int wave  = blockIdx.x * (blockDim.x >> 5) + (threadIdx.x >> 5);
    int nwav  = gridDim.x * (blockDim.x >> 5);
    float acc = 0.0f;
    for (int r = wave; r < n; r += nwav) acc += h[(size_t)r * 32 + lane];
    __hip_atomic_fetch_add(&g[lane], acc, __ATOMIC_RELAXED, __HIP_MEMORY_SCOPE_AGENT);
}

__global__ void head_kernel(const float* __restrict__ g,
                            const float* __restrict__ fc_w,
                            const float* __restrict__ fc_b,
                            float* __restrict__ out, float invN) {
    int j = threadIdx.x;
    if (j >= 10) return;
    float z = fc_b[j];
    for (int c = 0; c < 32; ++c) z = fmaf(g[c] * invN, fc_w[c * 10 + j], z);
    out[j] = 1.0f / (1.0f + expf(-z));
}

// ---------------------------------------------------------------------------
// Launch
// ---------------------------------------------------------------------------
extern "C" void kernel_launch(void* const* d_in, const int* in_sizes, int n_in,
                              void* d_out, int out_size, void* d_ws, size_t ws_size,
                              hipStream_t stream) {
    const float* x      = (const float*)d_in[0];
    const int*   ei     = (const int*)d_in[1];
    const float* enc_w1 = (const float*)d_in[2];
    const float* enc_b1 = (const float*)d_in[3];
    const float* enc_w2 = (const float*)d_in[4];
    const float* enc_b2 = (const float*)d_in[5];
    const float* enc_w3 = (const float*)d_in[6];
    const float* enc_b3 = (const float*)d_in[7];
    const float* c1_wl  = (const float*)d_in[8];
    const float* c1_bl  = (const float*)d_in[9];
    const float* c1_wr  = (const float*)d_in[10];
    const float* c2_wl  = (const float*)d_in[11];
    const float* c2_bl  = (const float*)d_in[12];
    const float* c2_wr  = (const float*)d_in[13];
    const float* c3_wl  = (const float*)d_in[14];
    const float* c3_bl  = (const float*)d_in[15];
    const float* c3_wr  = (const float*)d_in[16];
    const float* fc_w   = (const float*)d_in[17];
    const float* fc_b   = (const float*)d_in[18];
    float* out = (float*)d_out;

    const int N = in_sizes[0] / 2;       // 100000 (multiple of 16)
    const int E = in_sizes[1] / 2;       // 3200000

    // Workspace carve-out (256B aligned), ~90.5 MB total.
    char* p = (char*)d_ws;
    auto carve = [&](size_t bytes) {
        char* q = p;
        p += (bytes + 255) & ~(size_t)255;
        return q;
    };
    float*  h0   = (float*)carve((size_t)N * 32 * 4);
    float*  bufA = (float*)carve((size_t)N * 64 * 4);
    float*  bufB = (float*)carve((size_t)N * 64 * 4);
    float*  agg  = (float*)carve((size_t)N * 64 * 4);
    float*  cnt  = (float*)carve((size_t)N * 4);
    float*  g    = (float*)carve(32 * 4);
    __bf16* w1lf = (__bf16*)carve(32 * 64 * 2);
    __bf16* w1rf = (__bf16*)carve(32 * 64 * 2);
    __bf16* w2lf = (__bf16*)carve(64 * 64 * 2);
    __bf16* w2rf = (__bf16*)carve(64 * 64 * 2);
    __bf16* w3lf = (__bf16*)carve(64 * 32 * 2);
    __bf16* w3rf = (__bf16*)carve(64 * 32 * 2);

    // Per-call zero init (harness does not re-poison between replays).
    hipMemsetAsync(cnt, 0, (size_t)N * 4, stream);
    hipMemsetAsync(g, 0, 32 * 4, stream);

    // Pack weights into WMMA B-fragment order (bf16).
    pack_w_kernel<32, 64><<<2, 128, 0, stream>>>(c1_wl, w1lf);
    pack_w_kernel<32, 64><<<2, 128, 0, stream>>>(c1_wr, w1rf);
    pack_w_kernel<64, 64><<<2, 128, 0, stream>>>(c2_wl, w2lf);
    pack_w_kernel<64, 64><<<2, 128, 0, stream>>>(c2_wr, w2rf);
    pack_w_kernel<64, 32><<<1, 128, 0, stream>>>(c3_wl, w3lf);
    pack_w_kernel<64, 32><<<1, 128, 0, stream>>>(c3_wr, w3rf);

    // Encoder + degrees.
    encoder_kernel<<<(N + 255) / 256, 256, 0, stream>>>(
        x, enc_w1, enc_b1, enc_w2, enc_b2, enc_w3, enc_b3, h0, N);
    degree_kernel<<<(E + 255) / 256, 256, 0, stream>>>(ei, cnt, E);

    const int tiles   = N / 16;
    const int ublocks = (tiles + 3) / 4;   // 4 waves / 128-thread block

    // Layer 1: 32 -> 64
    hipMemsetAsync(agg, 0, (size_t)N * 32 * 4, stream);
    {
        long long t = (long long)E * 8;
        scatter_kernel<32><<<(int)((t + 255) / 256), 256, 0, stream>>>(ei, h0, agg, E);
    }
    sage_update_kernel<32, 64><<<ublocks, 128, 0, stream>>>(
        agg, cnt, h0, w1lf, c1_bl, w1rf, bufA, tiles);

    // Layer 2: 64 -> 64
    hipMemsetAsync(agg, 0, (size_t)N * 64 * 4, stream);
    {
        long long t = (long long)E * 16;
        scatter_kernel<64><<<(int)((t + 255) / 256), 256, 0, stream>>>(ei, bufA, agg, E);
    }
    sage_update_kernel<64, 64><<<ublocks, 128, 0, stream>>>(
        agg, cnt, bufA, w2lf, c2_bl, w2rf, bufB, tiles);

    // Layer 3: 64 -> 32
    hipMemsetAsync(agg, 0, (size_t)N * 64 * 4, stream);
    {
        long long t = (long long)E * 16;
        scatter_kernel<64><<<(int)((t + 255) / 256), 256, 0, stream>>>(ei, bufB, agg, E);
    }
    sage_update_kernel<64, 32><<<ublocks, 128, 0, stream>>>(
        agg, cnt, bufB, w3lf, c3_bl, w3rf, bufA, tiles);

    // Mean pool + classifier head.
    pool_kernel<<<512, 256, 0, stream>>>(bufA, g, N);
    head_kernel<<<1, 32, 0, stream>>>(g, fc_w, fc_b, out, 1.0f / (float)N);

    (void)n_in; (void)out_size; (void)ws_size;
}
